// LSTMSeq2SeqModel_15436112461863
// MI455X (gfx1250) — matrix operand order repaired and checked by reading, not compiled, not run on hardware
//
#include <hip/hip_runtime.h>
#include <stdint.h>

// ---------------- CDNA5 WMMA types ----------------
typedef __attribute__((ext_vector_type(16))) __bf16 v16bf;
typedef __attribute__((ext_vector_type(8)))  float  v8f;

union BFrag {
  v16bf v;
  uint4 q[2];
  unsigned short us[16];
};
static_assert(sizeof(BFrag) == 32, "frag must be 32B");

__device__ __forceinline__ unsigned short f2bf(float f) {
  unsigned u = __float_as_uint(f);
  u += 0x7FFFu + ((u >> 16) & 1u);          // RNE
  return (unsigned short)(u >> 16);
}
__device__ __forceinline__ float sigf(float x)  { return 1.0f / (1.0f + __expf(-x)); }
__device__ __forceinline__ float tanhx(float x) { return 2.0f / (1.0f + __expf(-2.0f * x)) - 1.0f; }
__device__ __forceinline__ v8f v8zero() {
  v8f z;
#pragma unroll
  for (int i = 0; i < 8; ++i) z[i] = 0.0f;
  return z;
}
__device__ __forceinline__ v8f wmma_bf16(const BFrag a, const BFrag b, v8f c) {
  // D = A(16x32 bf16) x B(32x16 bf16) + C(16x16 f32)
  return __builtin_amdgcn_wmma_f32_16x16x32_bf16(false, a.v, false, b.v, (short)0, c, false, false);
}

// ---------------- packing kernels ----------------
// dst layout: [Ntiles][Kc][lane 0..31][j 0..15] bf16, matching WMMA B-operand:
//   n = nt*16 + (lane&15), k = kc*32 + 8*(lane>>4) + j + 8*(j>>3)
__global__ void pack_weight(const float* __restrict__ W, int N, int K, int rowStride,
                            int colOff, unsigned short* __restrict__ dst,
                            int Ntiles, int Kc) {
  long tid = (long)blockIdx.x * blockDim.x + threadIdx.x;
  long total = (long)Ntiles * Kc * 512;
  if (tid >= total) return;
  int j    = (int)(tid & 15);
  int lane = (int)((tid >> 4) & 31);
  long blk = tid >> 9;                 // nt*Kc + kc
  int kc   = (int)(blk % Kc);
  int nt   = (int)(blk / Kc);
  int n = nt * 16 + (lane & 15);
  int k = kc * 32 + 8 * (lane >> 4) + j + 8 * (j >> 3);
  float v = 0.0f;
  if (n < N && k < K) v = W[(long)n * rowStride + colOff + k];
  dst[tid] = f2bf(v);
}

// inp[s][b][j] bf16, j: 0..15 = x, 16..23 = u, 24..31 = 0 (K pad to 32)
__global__ void pack_input(const float* __restrict__ x, const float* __restrict__ u,
                           unsigned short* __restrict__ dst, int S) {
  long tid = (long)blockIdx.x * blockDim.x + threadIdx.x;
  long total = (long)S * 16 * 32;
  if (tid >= total) return;
  int j = (int)(tid & 31);
  int b = (int)((tid >> 5) & 15);
  int s = (int)(tid >> 9);
  float v = 0.0f;
  if (j < 16)       v = x[((long)b * S + s) * 16 + j];
  else if (j < 24)  v = u[((long)b * S + s) * 8 + (j - 16)];
  dst[tid] = f2bf(v);
}

// ---------------- generic batched 16xNxK GEMM ----------------
// A: [S][16][Kc*32] bf16 row-major, B: packed, out = A@B.T(+bias)
// one wave per (s, nt) tile, 8 waves/block
__global__ __launch_bounds__(256) void gemm16(
    const unsigned short* __restrict__ A, const unsigned short* __restrict__ Bp,
    const float* __restrict__ bias0, const float* __restrict__ bias1,
    float* __restrict__ outSBN, float* __restrict__ outBSN,
    unsigned short* __restrict__ outBF,
    int S, int Ntiles, int Kc, int Nstr) {
  int lane = threadIdx.x & 31;
  int wave = threadIdx.x >> 5;
  long tile = (long)blockIdx.x * 8 + wave;
  long total = (long)S * Ntiles;
  if (tile >= total) return;
  int s  = (int)(tile / Ntiles);
  int nt = (int)(tile % Ntiles);
  int m = lane & 15, hf = lane >> 4;
  int Kpad = Kc * 32;
  v8f acc = v8zero();
  const unsigned short* Arow = A + ((long)s * 16 + m) * Kpad + 8 * hf;
  const unsigned short* Bb   = Bp + ((long)nt * Kc) * 512 + lane * 16;
  for (int kc = 0; kc < Kc; ++kc) {
    BFrag a, b;
    a.q[0] = *(const uint4*)(Arow + kc * 32);
    a.q[1] = *(const uint4*)(Arow + kc * 32 + 16);
    b.q[0] = *(const uint4*)(Bb + kc * 512);
    b.q[1] = *(const uint4*)(Bb + kc * 512 + 8);
    acc = wmma_bf16(a, b, acc);
  }
  int n = nt * 16 + m;
  float badd = 0.0f;
  if (bias0) badd += bias0[n];
  if (bias1) badd += bias1[n];
#pragma unroll
  for (int r = 0; r < 8; ++r) {
    int mm = r + 8 * hf;
    float v = acc[r] + badd;
    long idxS = ((long)s * 16 + mm) * Nstr + n;
    if (outSBN) outSBN[idxS] = v;
    if (outBSN) outBSN[((long)mm * S + s) * Nstr + n] = v;
    if (outBF)  outBF[idxS] = f2bf(v);
  }
}

// ---------------- encoder layer 0 (fused input proj, K=32) ----------------
// grid=2 (dir), block=512 (16 waves); wave w owns h cols [16w,16w+16)
__global__ __launch_bounds__(512) void lstm_l0(
    const unsigned short* __restrict__ inp,           // [S][16][32]
    const unsigned short* __restrict__ WihF, const unsigned short* __restrict__ WhhF,
    const float* __restrict__ bihF, const float* __restrict__ bhhF,
    const unsigned short* __restrict__ WihB, const unsigned short* __restrict__ WhhB,
    const float* __restrict__ bihB, const float* __restrict__ bhhB,
    unsigned short* __restrict__ out,                 // [S][16][512], cols dir*256..
    int S) {
  __shared__ unsigned short hl[16][256];
  int dir = blockIdx.x;
  const unsigned short* Wih = dir ? WihB : WihF;
  const unsigned short* Whh = dir ? WhhB : WhhF;
  const float* bih = dir ? bihB : bihF;
  const float* bhh = dir ? bhhB : bhhF;
  int colOff = dir ? 256 : 0;
  int lane = threadIdx.x & 31, wave = threadIdx.x >> 5;
  int m = lane & 15, hf = lane >> 4, ht = wave;
  for (int i = threadIdx.x; i < 16 * 256; i += 512) ((unsigned short*)hl)[i] = 0;
  float bs[4];
#pragma unroll
  for (int g = 0; g < 4; ++g) {
    int n = g * 256 + ht * 16 + m;
    bs[g] = bih[n] + bhh[n];
  }
  v8f c = v8zero();
  __syncthreads();
  for (int step = 0; step < S; ++step) {
    int t = dir ? (S - 1 - step) : step;
    v8f acc[4];
#pragma unroll
    for (int g = 0; g < 4; ++g) acc[g] = v8zero();
    { // input chunk (K=24 padded to 32)
      BFrag a;
      const unsigned short* ar = inp + ((long)t * 16 + m) * 32 + 8 * hf;
      a.q[0] = *(const uint4*)ar;
      a.q[1] = *(const uint4*)(ar + 16);
#pragma unroll
      for (int g = 0; g < 4; ++g) {
        BFrag b;
        const unsigned short* br = Wih + ((long)(g * 16 + ht)) * 512 + lane * 16;
        b.q[0] = *(const uint4*)br;
        b.q[1] = *(const uint4*)(br + 8);
        acc[g] = wmma_bf16(a, b, acc[g]);
      }
    }
    if (step + 1 < S) {
      int tn = dir ? (S - 2 - step) : (step + 1);
      __builtin_prefetch(inp + ((long)tn * 16 + m) * 32, 0, 1);
    }
    for (int kc = 0; kc < 8; ++kc) { // recurrent K=256
      BFrag a;
      const unsigned short* ar = &hl[m][kc * 32 + 8 * hf];
      a.q[0] = *(const uint4*)ar;
      a.q[1] = *(const uint4*)(ar + 16);
#pragma unroll
      for (int g = 0; g < 4; ++g) {
        BFrag b;
        const unsigned short* br = Whh + (((long)(g * 16 + ht)) * 8 + kc) * 512 + lane * 16;
        b.q[0] = *(const uint4*)br;
        b.q[1] = *(const uint4*)(br + 8);
        acc[g] = wmma_bf16(a, b, acc[g]);
      }
    }
    __syncthreads();  // all reads of hl done
#pragma unroll
    for (int r = 0; r < 8; ++r) {
      int mm = r + 8 * hf;
      float iv = acc[0][r] + bs[0], fv = acc[1][r] + bs[1];
      float gv = acc[2][r] + bs[2], ov = acc[3][r] + bs[3];
      float cv = sigf(fv) * c[r] + sigf(iv) * tanhx(gv);
      float hv = sigf(ov) * tanhx(cv);
      c[r] = cv;
      unsigned short hb = f2bf(hv);
      hl[mm][ht * 16 + m] = hb;
      out[((long)t * 16 + mm) * 512 + colOff + ht * 16 + m] = hb;
    }
    __syncthreads();  // writes visible before next-step reads
  }
}

// ---------------- encoder layer 1 (precomputed input proj) ----------------
__global__ __launch_bounds__(512) void lstm_l1(
    const float* __restrict__ XF, const float* __restrict__ XB, // [S][16][1024]
    const unsigned short* __restrict__ WhhF, const unsigned short* __restrict__ WhhB,
    unsigned short* __restrict__ out,                           // [S][16][512]
    float* __restrict__ hFin, float* __restrict__ cFin,         // dir==1 final state
    int S) {
  __shared__ unsigned short hl[16][256];
  int dir = blockIdx.x;
  const float* Xp = dir ? XB : XF;
  const unsigned short* Whh = dir ? WhhB : WhhF;
  int colOff = dir ? 256 : 0;
  int lane = threadIdx.x & 31, wave = threadIdx.x >> 5;
  int m = lane & 15, hf = lane >> 4, ht = wave;
  for (int i = threadIdx.x; i < 16 * 256; i += 512) ((unsigned short*)hl)[i] = 0;
  v8f c = v8zero();
  __syncthreads();
  for (int step = 0; step < S; ++step) {
    int t = dir ? (S - 1 - step) : step;
    v8f acc[4];
#pragma unroll
    for (int g = 0; g < 4; ++g) {
#pragma unroll
      for (int r = 0; r < 8; ++r)
        acc[g][r] = Xp[((long)t * 16 + (r + 8 * hf)) * 1024 + g * 256 + ht * 16 + m];
    }
    if (step + 1 < S) {
      int tn = dir ? (S - 2 - step) : (step + 1);
      __builtin_prefetch(&Xp[((long)tn * 16 + m) * 1024 + ht * 16], 0, 1);
    }
    for (int kc = 0; kc < 8; ++kc) {
      BFrag a;
      const unsigned short* ar = &hl[m][kc * 32 + 8 * hf];
      a.q[0] = *(const uint4*)ar;
      a.q[1] = *(const uint4*)(ar + 16);
#pragma unroll
      for (int g = 0; g < 4; ++g) {
        BFrag b;
        const unsigned short* br = Whh + (((long)(g * 16 + ht)) * 8 + kc) * 512 + lane * 16;
        b.q[0] = *(const uint4*)br;
        b.q[1] = *(const uint4*)(br + 8);
        acc[g] = wmma_bf16(a, b, acc[g]);
      }
    }
    __syncthreads();
#pragma unroll
    for (int r = 0; r < 8; ++r) {
      int mm = r + 8 * hf;
      float iv = acc[0][r], fv = acc[1][r], gv = acc[2][r], ov = acc[3][r];
      float cv = sigf(fv) * c[r] + sigf(iv) * tanhx(gv);
      float hv = sigf(ov) * tanhx(cv);
      c[r] = cv;
      unsigned short hb = f2bf(hv);
      hl[mm][ht * 16 + m] = hb;
      out[((long)t * 16 + mm) * 512 + colOff + ht * 16 + m] = hb;
      if (dir == 1 && step == S - 1) {
        hFin[mm * 256 + ht * 16 + m] = hv;
        cFin[mm * 256 + ht * 16 + m] = cv;
      }
    }
    __syncthreads();
  }
}

// ---------------- decoder (persistent single block, 16 waves) ----------------
__global__ __launch_bounds__(512) void decoder(
    const float* __restrict__ x_seq, const float* __restrict__ u_seq,
    const float* __restrict__ hbL, const float* __restrict__ cbL,
    const float* __restrict__ EA,   // [b][s][h] = enc_out@Wa2.T + ba
    const float* __restrict__ encf, // [b][s][h]
    const float* __restrict__ va,
    const unsigned short* __restrict__ Wa1p,
    const unsigned short* __restrict__ dWih0p, const unsigned short* __restrict__ dWhh0p,
    const float* __restrict__ dbih0, const float* __restrict__ dbhh0,
    const unsigned short* __restrict__ dWih1p, const unsigned short* __restrict__ dWhh1p,
    const float* __restrict__ dbih1, const float* __restrict__ dbhh1,
    const unsigned short* __restrict__ Wo1p, const float* __restrict__ bo1,
    const unsigned short* __restrict__ Wo2p, const float* __restrict__ bo2,
    float* __restrict__ out_us, float* __restrict__ out_ws,
    int S, int T) {
  __shared__ unsigned short h0b[16][256];
  __shared__ unsigned short h1b[16][256];
  __shared__ unsigned short din[16][288];   // [x(16) u(8) ctx(256) pad(8)]
  __shared__ unsigned short t1b[16][256];
  __shared__ float qproj[16][256];
  __shared__ float vas[256];

  int tid = threadIdx.x;
  int lane = tid & 31, wave = tid >> 5;
  int m = lane & 15, hf = lane >> 4, ht = wave;

  for (int i = tid; i < 16 * 256; i += 512) {
    unsigned short hb = f2bf(hbL[i]);
    ((unsigned short*)h0b)[i] = hb;
    ((unsigned short*)h1b)[i] = hb;
  }
  for (int i = tid; i < 256; i += 512) vas[i] = va[i];
  for (int i = tid; i < 16 * 288; i += 512) {
    int b = i / 288, j = i % 288;
    float v = 0.0f;
    if (j < 16)      v = x_seq[((long)b * S + (S - 1)) * 16 + j];
    else if (j < 24) v = u_seq[((long)b * S + (S - 1)) * 8 + (j - 16)];
    ((unsigned short*)din)[i] = f2bf(v);
  }
  v8f c0, c1;
#pragma unroll
  for (int r = 0; r < 8; ++r) {
    float cv = cbL[(r + 8 * hf) * 256 + ht * 16 + m];
    c0[r] = cv; c1[r] = cv;
  }
  float b0s[4], b1s[4];
#pragma unroll
  for (int g = 0; g < 4; ++g) {
    int n = g * 256 + ht * 16 + m;
    b0s[g] = dbih0[n] + dbhh0[n];
    b1s[g] = dbih1[n] + dbhh1[n];
  }
  __syncthreads();

  for (int t = 0; t < T; ++t) {
    // (a) qproj = h1 @ Wa1.T  (wave = one 16-col tile)
    {
      v8f acc = v8zero();
#pragma unroll
      for (int kc = 0; kc < 8; ++kc) {
        BFrag a, b;
        const unsigned short* ar = &h1b[m][kc * 32 + 8 * hf];
        a.q[0] = *(const uint4*)ar;
        a.q[1] = *(const uint4*)(ar + 16);
        const unsigned short* br = Wa1p + ((long)ht * 8 + kc) * 512 + lane * 16;
        b.q[0] = *(const uint4*)br;
        b.q[1] = *(const uint4*)(br + 8);
        acc = wmma_bf16(a, b, acc);
      }
#pragma unroll
      for (int r = 0; r < 8; ++r) qproj[r + 8 * hf][ht * 16 + m] = acc[r];
    }
    __syncthreads();

    // (b) scores + softmax; wave w owns batch row b=w (S=512: 16 s per lane)
    {
      int b = wave;
      float ev[16];
      float mx = -3.4e38f;
#pragma unroll
      for (int i = 0; i < 16; ++i) {
        int s = i * 32 + lane;
        const float* ea = EA + ((long)b * S + s) * 256;
        float sum = 0.0f;
        for (int h = 0; h < 256; ++h) sum += vas[h] * tanhx(ea[h] + qproj[b][h]);
        ev[i] = sum;
        mx = fmaxf(mx, sum);
      }
#pragma unroll
      for (int off = 16; off >= 1; off >>= 1) mx = fmaxf(mx, __shfl_xor(mx, off, 32));
      float den = 0.0f;
#pragma unroll
      for (int i = 0; i < 16; ++i) { ev[i] = __expf(ev[i] - mx); den += ev[i]; }
#pragma unroll
      for (int off = 16; off >= 1; off >>= 1) den += __shfl_xor(den, off, 32);
      float inv = 1.0f / den;
#pragma unroll
      for (int i = 0; i < 16; ++i) {
        int s = i * 32 + lane;
        out_ws[((long)b * T + t) * S + s] = ev[i] * inv;  // also serves as w buffer
      }
    }
    __syncthreads();

    // (c) ctx[b][h] = sum_s w[b][s]*enc[b][s][h]  (4096 pairs / 512 threads)
    for (int i = 0; i < 8; ++i) {
      int p = i * 512 + tid;
      int b = p >> 8, h = p & 255;
      const float* er = encf + ((long)b * S) * 256 + h;
      const float* wr = out_ws + ((long)b * T + t) * S;
      float sum = 0.0f;
      for (int s = 0; s < S; ++s) sum += wr[s] * er[(long)s * 256];
      din[b][24 + h] = f2bf(sum);
    }
    __syncthreads();

    // (d) cell0: g = din@dWih0.T + h0@dWhh0.T + b
    {
      v8f acc[4];
#pragma unroll
      for (int g = 0; g < 4; ++g) acc[g] = v8zero();
      for (int kc = 0; kc < 9; ++kc) {
        BFrag a;
        const unsigned short* ar = &din[m][kc * 32 + 8 * hf];
        a.q[0] = *(const uint4*)ar;
        a.q[1] = *(const uint4*)(ar + 16);
#pragma unroll
        for (int g = 0; g < 4; ++g) {
          BFrag b;
          const unsigned short* br = dWih0p + (((long)(g * 16 + ht)) * 9 + kc) * 512 + lane * 16;
          b.q[0] = *(const uint4*)br;
          b.q[1] = *(const uint4*)(br + 8);
          acc[g] = wmma_bf16(a, b, acc[g]);
        }
      }
      for (int kc = 0; kc < 8; ++kc) {
        BFrag a;
        const unsigned short* ar = &h0b[m][kc * 32 + 8 * hf];
        a.q[0] = *(const uint4*)ar;
        a.q[1] = *(const uint4*)(ar + 16);
#pragma unroll
        for (int g = 0; g < 4; ++g) {
          BFrag b;
          const unsigned short* br = dWhh0p + (((long)(g * 16 + ht)) * 8 + kc) * 512 + lane * 16;
          b.q[0] = *(const uint4*)br;
          b.q[1] = *(const uint4*)(br + 8);
          acc[g] = wmma_bf16(a, b, acc[g]);
        }
      }
      __syncthreads();  // reads of h0b done
#pragma unroll
      for (int r = 0; r < 8; ++r) {
        int mm = r + 8 * hf;
        float iv = acc[0][r] + b0s[0], fv = acc[1][r] + b0s[1];
        float gv = acc[2][r] + b0s[2], ov = acc[3][r] + b0s[3];
        float cv = sigf(fv) * c0[r] + sigf(iv) * tanhx(gv);
        float hv = sigf(ov) * tanhx(cv);
        c0[r] = cv;
        h0b[mm][ht * 16 + m] = f2bf(hv);
      }
    }
    __syncthreads();

    // (e) cell1: g = h0@dWih1.T + h1@dWhh1.T + b
    {
      v8f acc[4];
#pragma unroll
      for (int g = 0; g < 4; ++g) acc[g] = v8zero();
      for (int kc = 0; kc < 8; ++kc) {
        BFrag a;
        const unsigned short* ar = &h0b[m][kc * 32 + 8 * hf];
        a.q[0] = *(const uint4*)ar;
        a.q[1] = *(const uint4*)(ar + 16);
#pragma unroll
        for (int g = 0; g < 4; ++g) {
          BFrag b;
          const unsigned short* br = dWih1p + (((long)(g * 16 + ht)) * 8 + kc) * 512 + lane * 16;
          b.q[0] = *(const uint4*)br;
          b.q[1] = *(const uint4*)(br + 8);
          acc[g] = wmma_bf16(a, b, acc[g]);
        }
      }
      for (int kc = 0; kc < 8; ++kc) {
        BFrag a;
        const unsigned short* ar = &h1b[m][kc * 32 + 8 * hf];
        a.q[0] = *(const uint4*)ar;
        a.q[1] = *(const uint4*)(ar + 16);
#pragma unroll
        for (int g = 0; g < 4; ++g) {
          BFrag b;
          const unsigned short* br = dWhh1p + (((long)(g * 16 + ht)) * 8 + kc) * 512 + lane * 16;
          b.q[0] = *(const uint4*)br;
          b.q[1] = *(const uint4*)(br + 8);
          acc[g] = wmma_bf16(a, b, acc[g]);
        }
      }
      __syncthreads();
#pragma unroll
      for (int r = 0; r < 8; ++r) {
        int mm = r + 8 * hf;
        float iv = acc[0][r] + b1s[0], fv = acc[1][r] + b1s[1];
        float gv = acc[2][r] + b1s[2], ov = acc[3][r] + b1s[3];
        float cv = sigf(fv) * c1[r] + sigf(iv) * tanhx(gv);
        float hv = sigf(ov) * tanhx(cv);
        c1[r] = cv;
        h1b[mm][ht * 16 + m] = f2bf(hv);
      }
    }
    __syncthreads();

    // (f) t1 = relu(h1@Wo1.T + bo1)
    {
      v8f acc = v8zero();
#pragma unroll
      for (int kc = 0; kc < 8; ++kc) {
        BFrag a, b;
        const unsigned short* ar = &h1b[m][kc * 32 + 8 * hf];
        a.q[0] = *(const uint4*)ar;
        a.q[1] = *(const uint4*)(ar + 16);
        const unsigned short* br = Wo1p + ((long)ht * 8 + kc) * 512 + lane * 16;
        b.q[0] = *(const uint4*)br;
        b.q[1] = *(const uint4*)(br + 8);
        acc = wmma_bf16(a, b, acc);
      }
      float bo = bo1[ht * 16 + m];
#pragma unroll
      for (int r = 0; r < 8; ++r) {
        float v = fmaxf(acc[r] + bo, 0.0f);
        t1b[r + 8 * hf][ht * 16 + m] = f2bf(v);
      }
    }
    __syncthreads();

    // (g) u = t1@Wo2.T + bo2 (N=8 padded to one 16-tile; wave 0 only)
    if (wave == 0) {
      v8f acc = v8zero();
#pragma unroll
      for (int kc = 0; kc < 8; ++kc) {
        BFrag a, b;
        const unsigned short* ar = &t1b[m][kc * 32 + 8 * hf];
        a.q[0] = *(const uint4*)ar;
        a.q[1] = *(const uint4*)(ar + 16);
        const unsigned short* br = Wo2p + (long)kc * 512 + lane * 16;
        b.q[0] = *(const uint4*)br;
        b.q[1] = *(const uint4*)(br + 8);
        acc = wmma_bf16(a, b, acc);
      }
#pragma unroll
      for (int r = 0; r < 8; ++r) {
        int mm = r + 8 * hf;
        if (m < 8) {
          float uv = acc[r] + bo2[m];
          out_us[((long)mm * T + t) * 8 + m] = uv;
          din[mm][16 + m] = f2bf(uv);   // u feeds next decoder step
        }
      }
    }
    __syncthreads();
  }
}

// ---------------- host launcher ----------------
extern "C" void kernel_launch(void* const* d_in, const int* in_sizes, int n_in,
                              void* d_out, int out_size, void* d_ws, size_t ws_size,
                              hipStream_t stream) {
  const int B = 16, X = 16, U = 8;
  const float* x_seq = (const float*)d_in[0];
  const float* u_seq = (const float*)d_in[1];
  int S = in_sizes[0] / (B * X);                 // 512
  int T = out_size / (B * U + B * S);            // 48 (derived; d_in[2] is device scalar)

  const float* eWih_f0 = (const float*)d_in[3];
  const float* eWhh_f0 = (const float*)d_in[4];
  const float* ebih_f0 = (const float*)d_in[5];
  const float* ebhh_f0 = (const float*)d_in[6];
  const float* eWih_b0 = (const float*)d_in[7];
  const float* eWhh_b0 = (const float*)d_in[8];
  const float* ebih_b0 = (const float*)d_in[9];
  const float* ebhh_b0 = (const float*)d_in[10];
  const float* eWih_f1 = (const float*)d_in[11];
  const float* eWhh_f1 = (const float*)d_in[12];
  const float* ebih_f1 = (const float*)d_in[13];
  const float* ebhh_f1 = (const float*)d_in[14];
  const float* eWih_b1 = (const float*)d_in[15];
  const float* eWhh_b1 = (const float*)d_in[16];
  const float* ebih_b1 = (const float*)d_in[17];
  const float* ebhh_b1 = (const float*)d_in[18];
  const float* Wp      = (const float*)d_in[19];
  const float* bp      = (const float*)d_in[20];
  const float* dWih0   = (const float*)d_in[21];
  const float* dWhh0   = (const float*)d_in[22];
  const float* dbih0   = (const float*)d_in[23];
  const float* dbhh0   = (const float*)d_in[24];
  const float* dWih1   = (const float*)d_in[25];
  const float* dWhh1   = (const float*)d_in[26];
  const float* dbih1   = (const float*)d_in[27];
  const float* dbhh1   = (const float*)d_in[28];
  const float* Wa      = (const float*)d_in[29];
  const float* ba      = (const float*)d_in[30];
  const float* va      = (const float*)d_in[31];
  const float* Wo1     = (const float*)d_in[32];
  const float* bo1     = (const float*)d_in[33];
  const float* Wo2     = (const float*)d_in[34];
  const float* bo2     = (const float*)d_in[35];
  (void)n_in; (void)ws_size;

  char* p = (char*)d_ws;
  auto alloc = [&](size_t bytes) -> void* {
    void* r = (void*)p;
    p += (bytes + 255) & ~(size_t)255;
    return r;
  };
  auto alloc_us = [&](size_t n) { return (unsigned short*)alloc(n * 2); };
  auto alloc_f  = [&](size_t n) { return (float*)alloc(n * 4); };

  unsigned short* inp     = alloc_us((size_t)S * 16 * 32);
  unsigned short* Wih_f0p = alloc_us(64 * 1 * 512);
  unsigned short* Whh_f0p = alloc_us(64 * 8 * 512);
  unsigned short* Wih_b0p = alloc_us(64 * 1 * 512);
  unsigned short* Whh_b0p = alloc_us(64 * 8 * 512);
  unsigned short* Wih_f1p = alloc_us(64 * 16 * 512);
  unsigned short* Whh_f1p = alloc_us(64 * 8 * 512);
  unsigned short* Wih_b1p = alloc_us(64 * 16 * 512);
  unsigned short* Whh_b1p = alloc_us(64 * 8 * 512);
  unsigned short* Wpp     = alloc_us(16 * 16 * 512);
  unsigned short* dWih0p  = alloc_us(64 * 9 * 512);
  unsigned short* dWhh0p  = alloc_us(64 * 8 * 512);
  unsigned short* dWih1p  = alloc_us(64 * 8 * 512);
  unsigned short* dWhh1p  = alloc_us(64 * 8 * 512);
  unsigned short* Wa1p    = alloc_us(16 * 8 * 512);
  unsigned short* Wa2p    = alloc_us(16 * 8 * 512);
  unsigned short* Wo1p    = alloc_us(16 * 8 * 512);
  unsigned short* Wo2p    = alloc_us(1 * 8 * 512);
  unsigned short* l1out0  = alloc_us((size_t)S * 16 * 512);
  float*          Xf1     = alloc_f((size_t)S * 16 * 1024);
  float*          Xb1     = alloc_f((size_t)S * 16 * 1024);
  unsigned short* l2out   = alloc_us((size_t)S * 16 * 512);
  float*          encf    = alloc_f((size_t)16 * S * 256);
  unsigned short* encbf   = alloc_us((size_t)S * 16 * 256);
  float*          EA      = alloc_f((size_t)16 * S * 256);
  float*          hbL     = alloc_f(16 * 256);
  float*          cbL     = alloc_f(16 * 256);

  auto packW = [&](const float* W, int N, int K, int rs, int co, unsigned short* dst) {
    int Nt = (N + 15) / 16, Kc = (K + 31) / 32;
    long total = (long)Nt * Kc * 512;
    pack_weight<<<(unsigned)((total + 255) / 256), 256, 0, stream>>>(W, N, K, rs, co, dst, Nt, Kc);
  };

  {
    long tot = (long)S * 512;
    pack_input<<<(unsigned)((tot + 255) / 256), 256, 0, stream>>>(x_seq, u_seq, inp, S);
  }
  packW(eWih_f0, 1024, 24, 24, 0, Wih_f0p);
  packW(eWhh_f0, 1024, 256, 256, 0, Whh_f0p);
  packW(eWih_b0, 1024, 24, 24, 0, Wih_b0p);
  packW(eWhh_b0, 1024, 256, 256, 0, Whh_b0p);
  packW(eWih_f1, 1024, 512, 512, 0, Wih_f1p);
  packW(eWhh_f1, 1024, 256, 256, 0, Whh_f1p);
  packW(eWih_b1, 1024, 512, 512, 0, Wih_b1p);
  packW(eWhh_b1, 1024, 256, 256, 0, Whh_b1p);
  packW(Wp, 256, 512, 512, 0, Wpp);
  packW(dWih0, 1024, 280, 280, 0, dWih0p);
  packW(dWhh0, 1024, 256, 256, 0, dWhh0p);
  packW(dWih1, 1024, 256, 256, 0, dWih1p);
  packW(dWhh1, 1024, 256, 256, 0, dWhh1p);
  packW(Wa, 256, 256, 512, 0,   Wa1p);   // Wa[:, :H]  (q part)
  packW(Wa, 256, 256, 512, 256, Wa2p);   // Wa[:, H:]  (enc part, hoisted)
  packW(Wo1, 256, 256, 256, 0, Wo1p);
  packW(Wo2, 8, 256, 256, 0, Wo2p);

  // encoder layer 0 (both directions, one launch)
  lstm_l0<<<2, 512, 0, stream>>>(inp, Wih_f0p, Whh_f0p, ebih_f0, ebhh_f0,
                                 Wih_b0p, Whh_b0p, ebih_b0, ebhh_b0, l1out0, S);
  // batched layer-1 input projections (biases folded here)
  {
    long tiles = (long)S * 64;
    unsigned grid = (unsigned)((tiles + 7) / 8);
    gemm16<<<grid, 256, 0, stream>>>(l1out0, Wih_f1p, ebih_f1, ebhh_f1,
                                     Xf1, nullptr, nullptr, S, 64, 16, 1024);
    gemm16<<<grid, 256, 0, stream>>>(l1out0, Wih_b1p, ebih_b1, ebhh_b1,
                                     Xb1, nullptr, nullptr, S, 64, 16, 1024);
  }
  // encoder layer 1 (both directions) + final backward state capture
  lstm_l1<<<2, 512, 0, stream>>>(Xf1, Xb1, Whh_f1p, Whh_b1p, l2out, hbL, cbL, S);
  // enc_out = l2out @ Wp.T + bp ; EA = enc_out @ Wa2.T + ba (attention hoist)
  {
    long tiles = (long)S * 16;
    unsigned grid = (unsigned)((tiles + 7) / 8);
    gemm16<<<grid, 256, 0, stream>>>(l2out, Wpp, bp, nullptr,
                                     nullptr, encf, encbf, S, 16, 16, 256);
    gemm16<<<grid, 256, 0, stream>>>(encbf, Wa2p, ba, nullptr,
                                     nullptr, EA, nullptr, S, 16, 8, 256);
  }
  float* out_us = (float*)d_out;
  float* out_ws = out_us + (size_t)B * T * U;
  decoder<<<1, 512, 0, stream>>>(x_seq, u_seq, hbL, cbL, EA, encf, va,
                                 Wa1p, dWih0p, dWhh0p, dbih0, dbhh0,
                                 dWih1p, dWhh1p, dbih1, dbhh1,
                                 Wo1p, bo1, Wo2p, bo2, out_us, out_ws, S, T);
}
